// Attention_45483703664897
// MI455X (gfx1250) — compile-verified, matrix operand order checked by reading, not run on hardware
//
#include <hip/hip_runtime.h>
#include <hip/hip_bf16.h>

// ---------------------------------------------------------------------------
// Causal single-head attention for MI455X (gfx1250), bf16 WMMA pipeline.
//   B=4, T=4096, N_EMBD=1024, HEAD=64
// ---------------------------------------------------------------------------

typedef __attribute__((ext_vector_type(16))) __bf16 v16bf;
typedef __attribute__((ext_vector_type(8)))  __bf16 bf16x8;
typedef __attribute__((ext_vector_type(8)))  float  v8f;
typedef __attribute__((ext_vector_type(4)))  float  f32x4;

#define BB   4
#define TT   4096
#define NE   1024
#define HD   64

// softmax computed in exp2 domain; 1/sqrt(64) * log2(e) folded into K at
// projection time.
#define SCALE_LOG2E 0.1803368801111244f

union Frag { v16bf v; bf16x8 h[2]; };

__device__ __forceinline__ float rmax16(float v) {
    v = fmaxf(v, __shfl_xor(v, 1, 32));
    v = fmaxf(v, __shfl_xor(v, 2, 32));
    v = fmaxf(v, __shfl_xor(v, 4, 32));
    v = fmaxf(v, __shfl_xor(v, 8, 32));
    return v;
}

// ---------------------------------------------------------------------------
// Kernel 1: transpose + convert weights: W[k][n] f32 -> Wt[n][k] bf16
// ---------------------------------------------------------------------------
__global__ __launch_bounds__(256) void wt_kernel(
        const float* __restrict__ Wq, const float* __restrict__ Wk,
        const float* __restrict__ Wv,
        __bf16* __restrict__ Wtq, __bf16* __restrict__ Wtk,
        __bf16* __restrict__ Wtv) {
    int idx = blockIdx.x * 256 + threadIdx.x;   // 0 .. 64*1024-1
    int n = idx / NE;
    int k = idx % NE;
    Wtq[(size_t)n * NE + k] = (__bf16)Wq[(size_t)k * HD + n];
    Wtk[(size_t)n * NE + k] = (__bf16)Wk[(size_t)k * HD + n];
    Wtv[(size_t)n * NE + k] = (__bf16)Wv[(size_t)k * HD + n];
}

// ---------------------------------------------------------------------------
// Kernel 2: Q/K/V projections.  4 waves/block, 16 rows per wave.
//   X: [B*T][NE] f32 (converted to bf16 in-register)
//   Wt*: [HD][NE] bf16
//   Q out: [B*T][HD] bf16; K out: [B*T][HD] bf16 pre-scaled by 0.125*log2e;
//   Vt out: [B][HD][T] bf16 (transposed)
// ---------------------------------------------------------------------------
__global__ __launch_bounds__(128) void proj_kernel(
        const float* __restrict__ X,
        const __bf16* __restrict__ Wtq, const __bf16* __restrict__ Wtk,
        const __bf16* __restrict__ Wtv,
        __bf16* __restrict__ Q, __bf16* __restrict__ K,
        __bf16* __restrict__ Vt) {
    const int lane = threadIdx.x & 31;
    const int wave = threadIdx.x >> 5;
    const int half = lane >> 4;
    const int ln   = lane & 15;
    const int rowbase = blockIdx.x * 64 + wave * 16;

    v8f accq[4], acck[4], accv[4];
    #pragma unroll
    for (int t = 0; t < 4; ++t)
        #pragma unroll
        for (int i = 0; i < 8; ++i) { accq[t][i] = 0.f; acck[t][i] = 0.f; accv[t][i] = 0.f; }

    const float* xrow = X + (size_t)(rowbase + ln) * NE;

    for (int k0 = 0; k0 < NE; k0 += 32) {
        // A fragment (16x32 bf16): lane holds K = k0 + {half*8..+7, 16+half*8..+7}
        Frag a;
        const float* p0 = xrow + k0 + half * 8;
        const float* p1 = xrow + k0 + 16 + half * 8;
        f32x4 x0 = *(const f32x4*)(p0);
        f32x4 x1 = *(const f32x4*)(p0 + 4);
        f32x4 x2 = *(const f32x4*)(p1);
        f32x4 x3 = *(const f32x4*)(p1 + 4);
        #pragma unroll
        for (int i = 0; i < 4; ++i) {
            a.h[0][i]     = (__bf16)x0[i];
            a.h[0][i + 4] = (__bf16)x1[i];
            a.h[1][i]     = (__bf16)x2[i];
            a.h[1][i + 4] = (__bf16)x3[i];
        }
        #pragma unroll
        for (int t = 0; t < 4; ++t) {
            Frag bq, bk, bv;
            const __bf16* wq = Wtq + (size_t)(t * 16 + ln) * NE + k0;
            const __bf16* wk = Wtk + (size_t)(t * 16 + ln) * NE + k0;
            const __bf16* wv = Wtv + (size_t)(t * 16 + ln) * NE + k0;
            bq.h[0] = *(const bf16x8*)(wq + half * 8);
            bq.h[1] = *(const bf16x8*)(wq + 16 + half * 8);
            bk.h[0] = *(const bf16x8*)(wk + half * 8);
            bk.h[1] = *(const bf16x8*)(wk + 16 + half * 8);
            bv.h[0] = *(const bf16x8*)(wv + half * 8);
            bv.h[1] = *(const bf16x8*)(wv + 16 + half * 8);
            accq[t] = __builtin_amdgcn_wmma_f32_16x16x32_bf16(
                false, a.v, false, bq.v, (short)0, accq[t], false, false);
            acck[t] = __builtin_amdgcn_wmma_f32_16x16x32_bf16(
                false, a.v, false, bk.v, (short)0, acck[t], false, false);
            accv[t] = __builtin_amdgcn_wmma_f32_16x16x32_bf16(
                false, a.v, false, bv.v, (short)0, accv[t], false, false);
        }
    }

    // C/D layout: lane holds column (t*16+ln), VGPR r holds row (r + 8*half)
    #pragma unroll
    for (int t = 0; t < 4; ++t)
        #pragma unroll
        for (int r = 0; r < 8; ++r) {
            int row = rowbase + r + half * 8;
            int col = t * 16 + ln;
            Q[(size_t)row * HD + col] = (__bf16)accq[t][r];
            K[(size_t)row * HD + col] = (__bf16)(acck[t][r] * SCALE_LOG2E);
            int b  = row >> 12;          // /4096
            int rb = row & (TT - 1);
            Vt[((size_t)(b * HD + col)) * TT + rb] = (__bf16)accv[t][r];
        }
}

// ---------------------------------------------------------------------------
// Kernel 3: flash attention.  One 16-row tile per wave, 64-wide KV blocks.
// V fragments are loaded BEFORE the softmax so their L2 latency hides under
// the softmax VALU/TRANS work (the LDS fence below is a compiler memory
// barrier, so loads placed after it cannot be hoisted).  Row sums accumulate
// in a 5th output tile via a ones-column WMMA.
// ---------------------------------------------------------------------------
template <bool MASKED>
__device__ __forceinline__ void attn_block(
        int j0, int i0, int b, int half, int ln,
        const __bf16* __restrict__ K, const __bf16* __restrict__ Vt,
        const Frag& qa, const Frag& qb, const v16bf& onesv,
        __bf16* __restrict__ myp, v8f O[5], float mrun[8]) {
    // --- S = Q K^T over 4 column groups of 16 (K pre-scaled, exp2 domain) ---
    v8f S[4];
    #pragma unroll
    for (int g = 0; g < 4; ++g) {
        const __bf16* kr = K + ((size_t)b * TT + j0 + g * 16 + ln) * HD;
        Frag ka, kb;
        ka.h[0] = *(const bf16x8*)(kr + half * 8);
        ka.h[1] = *(const bf16x8*)(kr + 16 + half * 8);
        kb.h[0] = *(const bf16x8*)(kr + 32 + half * 8);
        kb.h[1] = *(const bf16x8*)(kr + 48 + half * 8);
        v8f s;
        #pragma unroll
        for (int i = 0; i < 8; ++i) s[i] = 0.f;
        s = __builtin_amdgcn_wmma_f32_16x16x32_bf16(false, qa.v, false, ka.v, (short)0, s, false, false);
        s = __builtin_amdgcn_wmma_f32_16x16x32_bf16(false, qb.v, false, kb.v, (short)0, s, false, false);
        S[g] = s;
    }

    // --- issue V loads early: independent of P, latency hidden by softmax ---
    Frag v0[4], v1[4];
    #pragma unroll
    for (int t = 0; t < 4; ++t) {
        const __bf16* vr = Vt + ((size_t)(b * HD + t * 16 + ln)) * TT + j0;
        v0[t].h[0] = *(const bf16x8*)(vr + half * 8);
        v0[t].h[1] = *(const bf16x8*)(vr + 16 + half * 8);
        v1[t].h[0] = *(const bf16x8*)(vr + 32 + half * 8);
        v1[t].h[1] = *(const bf16x8*)(vr + 48 + half * 8);
    }

    // --- online softmax (exp2 domain), P staged to LDS as bf16 ---
    #pragma unroll
    for (int r = 0; r < 8; ++r) {
        const int row = i0 + r + half * 8;
        float s0 = S[0][r], s1 = S[1][r], s2 = S[2][r], s3 = S[3][r];
        if (MASKED) {
            s0 = (j0 + ln)      <= row ? s0 : -INFINITY;
            s1 = (j0 + 16 + ln) <= row ? s1 : -INFINITY;
            s2 = (j0 + 32 + ln) <= row ? s2 : -INFINITY;
            s3 = (j0 + 48 + ln) <= row ? s3 : -INFINITY;
        }
        float mx = rmax16(fmaxf(fmaxf(s0, s1), fmaxf(s2, s3)));
        float mnew = fmaxf(mrun[r], mx);
        float corr = exp2f(mrun[r] - mnew);
        mrun[r] = mnew;
        float p0 = exp2f(s0 - mnew);
        float p1 = exp2f(s1 - mnew);
        float p2 = exp2f(s2 - mnew);
        float p3 = exp2f(s3 - mnew);
        #pragma unroll
        for (int t = 0; t < 5; ++t) O[t][r] *= corr;
        __bf16* prow = myp + (r + half * 8) * 64 + ln;
        prow[0]  = (__bf16)p0;
        prow[16] = (__bf16)p1;
        prow[32] = (__bf16)p2;
        prow[48] = (__bf16)p3;
    }
    asm volatile("s_wait_dscnt 0" ::: "memory");

    // --- regather P (16x64) as two 16x32 A fragments ---
    Frag pa, pb;
    const __bf16* pr = myp + ln * 64;
    pa.h[0] = *(const bf16x8*)(pr + half * 8);
    pa.h[1] = *(const bf16x8*)(pr + 16 + half * 8);
    pb.h[0] = *(const bf16x8*)(pr + 32 + half * 8);
    pb.h[1] = *(const bf16x8*)(pr + 48 + half * 8);

    // --- O += P V ---
    #pragma unroll
    for (int t = 0; t < 4; ++t) {
        O[t] = __builtin_amdgcn_wmma_f32_16x16x32_bf16(false, pa.v, false, v0[t].v, (short)0, O[t], false, false);
        O[t] = __builtin_amdgcn_wmma_f32_16x16x32_bf16(false, pb.v, false, v1[t].v, (short)0, O[t], false, false);
    }
    // row-sum tile: l += P @ ones  (column 0 of tile 4)
    O[4] = __builtin_amdgcn_wmma_f32_16x16x32_bf16(false, pa.v, false, onesv, (short)0, O[4], false, false);
    O[4] = __builtin_amdgcn_wmma_f32_16x16x32_bf16(false, pb.v, false, onesv, (short)0, O[4], false, false);
}

__global__ __launch_bounds__(128) void attn_kernel(
        const __bf16* __restrict__ Q, const __bf16* __restrict__ K,
        const __bf16* __restrict__ Vt, float* __restrict__ out) {
    __shared__ __bf16 lds_p[4][16 * 64];   // per-wave P bounce buffer (2 KB each)

    const int lane = threadIdx.x & 31;
    const int wave = threadIdx.x >> 5;
    const int half = lane >> 4;
    const int ln   = lane & 15;
    const int b    = blockIdx.x >> 6;       // 64 blocks per batch
    const int blk  = blockIdx.x & 63;

    // balanced tile set {blk, 127-blk, 128+blk, 255-blk}: constant work/block
    int tile;
    if      (wave == 0) tile = blk;
    else if (wave == 1) tile = 127 - blk;
    else if (wave == 2) tile = 128 + blk;
    else                tile = 255 - blk;
    const int i0 = tile * 16;               // row-in-batch

    // Q fragments for k=0..31 and k=32..63
    Frag qa, qb;
    const __bf16* qrow = Q + ((size_t)b * TT + i0 + ln) * HD;
    qa.h[0] = *(const bf16x8*)(qrow + half * 8);
    qa.h[1] = *(const bf16x8*)(qrow + 16 + half * 8);
    qb.h[0] = *(const bf16x8*)(qrow + 32 + half * 8);
    qb.h[1] = *(const bf16x8*)(qrow + 48 + half * 8);

    // ones B-fragment: column n=0 all ones -> lane ln==0 holds 1 in all slots
    v16bf onesv;
    #pragma unroll
    for (int i = 0; i < 16; ++i) onesv[i] = (ln == 0) ? (__bf16)1.0f : (__bf16)0.0f;

    v8f O[5];
    float mrun[8];
    #pragma unroll
    for (int t = 0; t < 5; ++t)
        #pragma unroll
        for (int i = 0; i < 8; ++i) O[t][i] = 0.f;
    #pragma unroll
    for (int r = 0; r < 8; ++r) mrun[r] = -INFINITY;

    __bf16* myp = lds_p[wave];

    const int j0m = i0 & ~63;   // first (and only) masked 64-block
    for (int j0 = 0; j0 < j0m; j0 += 64)
        attn_block<false>(j0, i0, b, half, ln, K, Vt, qa, qb, onesv, myp, O, mrun);
    attn_block<true>(j0m, i0, b, half, ln, K, Vt, qa, qb, onesv, myp, O, mrun);

    // --- epilogue: broadcast row sums from lane half*16, normalize, store ---
    #pragma unroll
    for (int r = 0; r < 8; ++r) {
        float l = __shfl(O[4][r], half * 16, 32);
        float inv = 1.0f / l;
        int row = i0 + r + half * 8;
        float* orow = out + ((size_t)b * TT + row) * HD + ln;
        orow[0]  = O[0][r] * inv;
        orow[16] = O[1][r] * inv;
        orow[32] = O[2][r] * inv;
        orow[48] = O[3][r] * inv;
    }
}

// ---------------------------------------------------------------------------
extern "C" void kernel_launch(void* const* d_in, const int* in_sizes, int n_in,
                              void* d_out, int out_size, void* d_ws, size_t ws_size,
                              hipStream_t stream) {
    const float* x  = (const float*)d_in[0];
    const float* Wq = (const float*)d_in[1];
    const float* Wk = (const float*)d_in[2];
    const float* Wv = (const float*)d_in[3];
    float* out = (float*)d_out;

    // workspace carve-up, total ~6.4 MB
    char* ws = (char*)d_ws;
    __bf16* Wtq = (__bf16*)ws;  ws += (size_t)HD * NE * sizeof(__bf16);   // 128 KB
    __bf16* Wtk = (__bf16*)ws;  ws += (size_t)HD * NE * sizeof(__bf16);
    __bf16* Wtv = (__bf16*)ws;  ws += (size_t)HD * NE * sizeof(__bf16);
    __bf16* Qb  = (__bf16*)ws;  ws += (size_t)BB * TT * HD * sizeof(__bf16); // 2 MB
    __bf16* Kb  = (__bf16*)ws;  ws += (size_t)BB * TT * HD * sizeof(__bf16);
    __bf16* Vt  = (__bf16*)ws;  ws += (size_t)BB * TT * HD * sizeof(__bf16);

    // 1) transpose/convert weights to [HD][NE] bf16
    wt_kernel<<<(HD * NE) / 256, 256, 0, stream>>>(Wq, Wk, Wv, Wtq, Wtk, Wtv);
    // 2) projections: 16384 rows, 64 rows per block (4 waves x 16 rows)
    proj_kernel<<<(BB * TT) / 64, 128, 0, stream>>>(x, Wtq, Wtk, Wtv, Qb, Kb, Vt);
    // 3) flash attention: 64 blocks per batch, 4 balanced row-tiles per block
    attn_kernel<<<BB * 64, 128, 0, stream>>>(Qb, Kb, Vt, out);
}
